// MambaLayersCenter_2319282340141
// MI455X (gfx1250) — matrix operand order repaired and checked by reading, not compiled, not run on hardware
//
#include <hip/hip_runtime.h>

#define BATCH 256
#define DM    256            // d_model
#define TLEN  169            // spiral sequence length (13*13)
#define DI    512            // d_inner
#define NS    16             // d_state
#define RNK   16             // dt_rank
#define MROWS (BATCH * TLEN) // 43264 = 32 * 1352

// ---------------- spiral index table (compile-time) ----------------
struct SpiralIdx { int v[TLEN]; };
static constexpr SpiralIdx make_spiral() {
    SpiralIdx s{};
    bool vis[13][13] = {};
    const int dx[4] = {0, 1, 0, -1};
    const int dy[4] = {1, 0, -1, 0};
    int x = 6, y = 6, step = 1, d = 0, n = 0;
    while (n < TLEN) {
        for (int r = 0; r < 2; ++r) {
            for (int k = 0; k < step; ++k) {
                if (x >= 0 && x < 13 && y >= 0 && y < 13 && !vis[x][y] && n < TLEN) {
                    vis[x][y] = true;
                    s.v[n++] = x * 13 + y;
                }
                x += dx[d]; y += dy[d];
            }
            d = (d + 1) & 3;
        }
        ++step;
    }
    return s;
}
__constant__ SpiralIdx c_spiral = make_spiral();

// ---------------- bf16 helpers ----------------
typedef __bf16 v16bf __attribute__((ext_vector_type(16)));
typedef float  v8f   __attribute__((ext_vector_type(8)));

__device__ __forceinline__ unsigned short f2bf(float f) {
    unsigned u = __float_as_uint(f);
    unsigned r = u + 0x7FFFu + ((u >> 16) & 1u);  // RNE
    return (unsigned short)(r >> 16);
}
__device__ __forceinline__ float bf2f(unsigned short h) {
    return __uint_as_float(((unsigned)h) << 16);
}

union FragU { v16bf v; unsigned short u[16]; };

// A fragment: 16x32 bf16, row-major source, lda = K.
// lanes 0-15: M=lane, K = [k..k+7] and [k+16..k+23]
// lanes 16-31: M=lane-16, K = [k+8..k+15] and [k+24..k+31]
__device__ __forceinline__ v16bf load_frag_a(const unsigned short* __restrict__ A,
                                             int K, int mbase, int k, int lane) {
    int m = mbase + (lane & 15);
    int koff = k + ((lane & 16) ? 8 : 0);
    const unsigned short* p = A + (size_t)m * K + koff;
    FragU f;
#pragma unroll
    for (int i = 0; i < 8; ++i) { f.u[i] = p[i]; f.u[8 + i] = p[16 + i]; }
    return f.v;
}

// B fragment: 32x16 bf16 (KxN), weight stored row-major as W[n][k] (ldw = K).
// lanes 0-15: N=lane, K=[k..k+7] and [k+16..k+23]; lanes 16-31: N=lane-16, K=[k+8..k+15] and [k+24..k+31]
__device__ __forceinline__ v16bf load_frag_b(const unsigned short* __restrict__ W,
                                             int K, int nbase, int k, int lane) {
    int n = nbase + (lane & 15);
    int koff = k + ((lane & 16) ? 8 : 0);
    const unsigned short* p = W + (size_t)n * K + koff;
    FragU f;
#pragma unroll
    for (int i = 0; i < 8; ++i) { f.u[i] = p[i]; f.u[8 + i] = p[16 + i]; }
    return f.v;
}

// ---------------- WMMA GEMM: C[M,N] = A[M,K] * W[N,K]^T (+ Cin) ----------------
// Each wave computes a 16x64 strip: A fragment loaded once per K-step, reused by
// 4 independent WMMAs (4 accumulators -> latency hiding, 1.25KB traffic per wmma).
// Block = 256 threads = 8 waves arranged 2(M) x 4(N); block tile = 32 x 256.
// Cin (fp32, ldc=N) optionally seeds the accumulators (fused residual add).
__global__ void __launch_bounds__(256)
wmma_gemm_bf16(const unsigned short* __restrict__ A,
               const unsigned short* __restrict__ W,
               const float* __restrict__ Cin,
               float* __restrict__ outF,
               unsigned short* __restrict__ outBF,
               int M, int N, int K) {
    int lane   = threadIdx.x & 31;
    int wave   = threadIdx.x >> 5;
    int mtile  = blockIdx.x * 32 + (wave & 1) * 16;
    int ntile0 = blockIdx.y * 256 + (wave >> 1) * 64;
    if (mtile >= M || ntile0 >= N) return;

    int nlo = lane & 15;                       // C/D: lane picks column
    int mo  = mtile + ((lane & 16) ? 8 : 0);   // lanes16-31 hold rows M+8..M+15

    v8f acc[4];
#pragma unroll
    for (int j = 0; j < 4; ++j) {
        int n = ntile0 + j * 16 + nlo;
        if (Cin) {
#pragma unroll
            for (int r = 0; r < 8; ++r) acc[j][r] = Cin[(size_t)(mo + r) * N + n];
        } else {
#pragma unroll
            for (int r = 0; r < 8; ++r) acc[j][r] = 0.0f;
        }
    }

    for (int k = 0; k < K; k += 32) {
        v16bf a = load_frag_a(A, K, mtile, k, lane);
#pragma unroll
        for (int j = 0; j < 4; ++j) {
            v16bf b = load_frag_b(W, K, ntile0 + j * 16, k, lane);
            acc[j] = __builtin_amdgcn_wmma_f32_16x16x32_bf16(
                /*neg_a=*/false, a, /*neg_b=*/false, b,
                /*c_mod=*/(short)0, acc[j], /*reuse_a=*/false, /*reuse_b=*/false);
        }
    }

#pragma unroll
    for (int j = 0; j < 4; ++j) {
        int n = ntile0 + j * 16 + nlo;
        if (outF) {
#pragma unroll
            for (int r = 0; r < 8; ++r) outF[(size_t)(mo + r) * N + n] = acc[j][r];
        } else {
#pragma unroll
            for (int r = 0; r < 8; ++r) outBF[(size_t)(mo + r) * N + n] = f2bf(acc[j][r]);
        }
    }
}

// ---------------- spiral gather: x[B,C,169] -> u[B,T,C] ----------------
__global__ void gather_kernel(const float* __restrict__ x, float* __restrict__ u) {
    int idx = blockIdx.x * blockDim.x + threadIdx.x;
    if (idx >= MROWS * DM) return;
    int c = idx % DM;
    int l = (idx / DM) % TLEN;
    int b = idx / (DM * TLEN);
    u[idx] = x[((size_t)b * DM + c) * TLEN + c_spiral.v[l]];
}

// ---------------- rmsnorm + bf16 pack ----------------
__global__ void rmsnorm_bf16_kernel(const float* __restrict__ u,
                                    const float* __restrict__ w,
                                    unsigned short* __restrict__ out) {
    int row = blockIdx.x;
    int c   = threadIdx.x;            // 256 threads = 8 waves
    float v = u[(size_t)row * DM + c];
    float s = v * v;
#pragma unroll
    for (int off = 16; off > 0; off >>= 1) s += __shfl_xor(s, off, 32);
    __shared__ float red[8];
    int lane = c & 31, wave = c >> 5;
    if (lane == 0) red[wave] = s;
    __syncthreads();
    if (wave == 0) {
        float t = (lane < 8) ? red[lane] : 0.0f;
#pragma unroll
        for (int off = 4; off > 0; off >>= 1) t += __shfl_xor(t, off, 32);
        if (lane == 0) red[0] = t;
    }
    __syncthreads();
    float r = rsqrtf(red[0] / (float)DM + 1e-5f);
    out[(size_t)row * DM + c] = f2bf(v * r * w[c]);
}

// ---------------- causal depthwise conv (d_conv=4) + silu ----------------
__global__ void conv_silu_kernel(const unsigned short* __restrict__ xz,
                                 const float* __restrict__ cw,
                                 const float* __restrict__ cb,
                                 float* __restrict__ xconv) {
    int idx = blockIdx.x * blockDim.x + threadIdx.x;  // (b*T+t)*DI + d
    if (idx >= MROWS * DI) return;
    int d = idx % DI;
    int t = (idx / DI) % TLEN;
    int b = idx / (DI * TLEN);
    float acc = cb[d];
#pragma unroll
    for (int j = 0; j < 4; ++j) {
        int tt = t - 3 + j;
        if (tt >= 0)
            acc += cw[d * 4 + j] * bf2f(xz[((size_t)(b * TLEN + tt)) * (2 * DI) + d]);
    }
    float sig = 1.0f / (1.0f + __expf(-acc));
    xconv[idx] = acc * sig;
}

// ---------------- x_proj: dbc[row, j<48] = xconv[row,:] . xw[j,:] ----------------
__global__ void xproj_kernel(const float* __restrict__ xconv,
                             const float* __restrict__ xw,
                             float* __restrict__ dbc) {
    int row = blockIdx.x;
    int j   = threadIdx.x;
    if (j >= 48) return;
    const float* xr = xconv + (size_t)row * DI;
    const float* wr = xw + (size_t)j * DI;
    float acc = 0.0f;
#pragma unroll 8
    for (int d = 0; d < DI; ++d) acc += xr[d] * wr[d];
    dbc[(size_t)row * 48 + j] = acc;
}

// ---------------- delta = softplus(dt @ dtw^T + dtb) ----------------
__global__ void delta_kernel(const float* __restrict__ dbc,
                             const float* __restrict__ dtw,
                             const float* __restrict__ dtb,
                             float* __restrict__ delta) {
    int idx = blockIdx.x * blockDim.x + threadIdx.x;  // row*DI + d
    if (idx >= MROWS * DI) return;
    int d   = idx % DI;
    int row = idx / DI;
    const float* dt = dbc + (size_t)row * 48;
    const float* w  = dtw + (size_t)d * RNK;
    float acc = dtb[d];
#pragma unroll
    for (int r = 0; r < RNK; ++r) acc += dt[r] * w[r];
    delta[idx] = (acc > 20.0f) ? acc : __logf(1.0f + __expf(acc));
}

// ---------------- selective scan; fused D*x and silu(z) gating ----------------
__global__ void __launch_bounds__(DI)
scan_kernel(const float* __restrict__ xconv,
            const float* __restrict__ delta,
            const float* __restrict__ dbc,
            const unsigned short* __restrict__ xz,
            const float* __restrict__ A_log,
            const float* __restrict__ Dp,
            unsigned short* __restrict__ yact) {
    int b = blockIdx.x;
    int d = threadIdx.x;  // 512 channels
    float A[NS], h[NS];
#pragma unroll
    for (int n = 0; n < NS; ++n) {
        A[n] = -__expf(A_log[(size_t)d * NS + n]);
        h[n] = 0.0f;
    }
    float Dd = Dp[d];
    __shared__ float sB[NS], sC[NS];
    for (int t = 0; t < TLEN; ++t) {
        size_t row = (size_t)b * TLEN + t;
        __syncthreads();
        if (d < 2 * NS) {
            if (d < NS) sB[d] = dbc[row * 48 + RNK + d];
            else        sC[d - NS] = dbc[row * 48 + RNK + NS + (d - NS)];
        }
        __syncthreads();
        float dt = delta[row * DI + d];
        float xv = xconv[row * DI + d];
        float dx = dt * xv;
        float y  = 0.0f;
#pragma unroll
        for (int n = 0; n < NS; ++n) {
            float dA = __expf(dt * A[n]);
            h[n] = dA * h[n] + dx * sB[n];
            y += h[n] * sC[n];
        }
        float zv  = bf2f(xz[row * (2 * DI) + DI + d]);
        float sig = 1.0f / (1.0f + __expf(-zv));
        yact[row * DI + d] = f2bf((y + Dd * xv) * (zv * sig));
    }
}

// ---------------- fp32 -> bf16 weight pack ----------------
__global__ void f2bf_kernel(const float* __restrict__ in,
                            unsigned short* __restrict__ out, int n) {
    int i = blockIdx.x * blockDim.x + threadIdx.x;
    if (i < n) out[i] = f2bf(in[i]);
}

// ---------------- u[B,T,C] -> out[B,C,T] ----------------
__global__ void final_kernel(const float* __restrict__ u, float* __restrict__ out) {
    int idx = blockIdx.x * blockDim.x + threadIdx.x;  // b*C*T + c*T + l
    if (idx >= MROWS * DM) return;
    int l = idx % TLEN;
    int c = (idx / TLEN) % DM;
    int b = idx / (TLEN * DM);
    out[idx] = u[((size_t)b * TLEN + l) * DM + c];
}

extern "C" void kernel_launch(void* const* d_in, const int* in_sizes, int n_in,
                              void* d_out, int out_size, void* d_ws, size_t ws_size,
                              hipStream_t stream) {
    (void)in_sizes; (void)n_in; (void)out_size; (void)ws_size;
    const float* x_in    = (const float*)d_in[0];
    const float* norm_w  = (const float*)d_in[1];
    const float* in_w    = (const float*)d_in[2];
    const float* conv_w  = (const float*)d_in[3];
    const float* conv_b  = (const float*)d_in[4];
    const float* xproj_w = (const float*)d_in[5];
    const float* dt_w    = (const float*)d_in[6];
    const float* dt_b    = (const float*)d_in[7];
    const float* A_log   = (const float*)d_in[8];
    const float* Dparam  = (const float*)d_in[9];
    const float* out_w   = (const float*)d_in[10];

    char* ws = (char*)d_ws;
    size_t off = 0;
    auto bump = [&](size_t bytes) -> void* {
        void* p = ws + off;
        off = (off + bytes + 255) & ~(size_t)255;
        return p;
    };
    float*          u_f     = (float*)bump((size_t)MROWS * DM * 4);
    unsigned short* un_bf   = (unsigned short*)bump((size_t)MROWS * DM * 2);
    unsigned short* xz_bf   = (unsigned short*)bump((size_t)MROWS * 2 * DI * 2);
    float*          xc_f    = (float*)bump((size_t)MROWS * DI * 4);
    float*          dbc_f   = (float*)bump((size_t)MROWS * 48 * 4);
    float*          delta_f = (float*)bump((size_t)MROWS * DI * 4);
    unsigned short* yact_bf = (unsigned short*)bump((size_t)MROWS * DI * 2);
    unsigned short* win_bf  = (unsigned short*)bump((size_t)2 * DI * DM * 2);
    unsigned short* wout_bf = (unsigned short*)bump((size_t)DM * DI * 2);

    gather_kernel<<<(MROWS * DM + 255) / 256, 256, 0, stream>>>(x_in, u_f);

    for (int l = 0; l < 2; ++l) {
        const float* lw_in  = in_w  + (size_t)l * 2 * DI * DM;
        const float* lw_out = out_w + (size_t)l * DM * DI;

        f2bf_kernel<<<(2 * DI * DM + 255) / 256, 256, 0, stream>>>(lw_in, win_bf, 2 * DI * DM);
        f2bf_kernel<<<(DM * DI + 255) / 256, 256, 0, stream>>>(lw_out, wout_bf, DM * DI);

        rmsnorm_bf16_kernel<<<MROWS, DM, 0, stream>>>(u_f, norm_w + (size_t)l * DM, un_bf);

        // xz[M, 2*DI] = un[M, DM] @ in_w^T    (bf16 WMMA, bf16 output)
        dim3 g1(MROWS / 32, (2 * DI) / 256);
        wmma_gemm_bf16<<<g1, 256, 0, stream>>>(un_bf, win_bf, nullptr, nullptr, xz_bf,
                                               MROWS, 2 * DI, DM);

        conv_silu_kernel<<<(MROWS * DI + 255) / 256, 256, 0, stream>>>(
            xz_bf, conv_w + (size_t)l * DI * 4, conv_b + (size_t)l * DI, xc_f);

        xproj_kernel<<<MROWS, 64, 0, stream>>>(xc_f, xproj_w + (size_t)l * 48 * DI, dbc_f);

        delta_kernel<<<(MROWS * DI + 255) / 256, 256, 0, stream>>>(
            dbc_f, dt_w + (size_t)l * DI * RNK, dt_b + (size_t)l * DI, delta_f);

        scan_kernel<<<BATCH, DI, 0, stream>>>(xc_f, delta_f, dbc_f, xz_bf,
                                              A_log + (size_t)l * DI * NS,
                                              Dparam + (size_t)l * DI, yact_bf);

        // u[M, DM] += yact[M, DI] @ out_w^T   (bf16 WMMA, fused residual via Cin, fp32 out)
        dim3 g2(MROWS / 32, DM / 256);
        wmma_gemm_bf16<<<g2, 256, 0, stream>>>(yact_bf, wout_bf, u_f, u_f, nullptr,
                                               MROWS, DM, DI);
    }

    final_kernel<<<(MROWS * DM + 255) / 256, 256, 0, stream>>>(u_f, (float*)d_out);
}
